// CrossEntropyGroup_18769007083970
// MI455X (gfx1250) — matrix operand order repaired and checked by reading, not compiled, not run on hardware
//
#include <hip/hip_runtime.h>

#define NUM_PROJ    128
#define NUM_GROUPS  64
#define IN_DIM      16384
#define NUM_PROTOS  10000
#define NUM_CLASSES 1000
#define NUM_PG      (NUM_PROJ * NUM_GROUPS)   // 8192
#define EPSV        1e-5f

#define KC      64              // K-chunk staged in LDS per iteration
#define LDST    72              // bf16 elems per LDS row (64 + 8 pad; 16B align kept)
#define CHELEM  (NUM_GROUPS * LDST)
#define KSPLIT  4               // K-segments per projection (blocks along K)
#define KPER    (IN_DIM / KSPLIT)   // 4096
#define NCHUNK  (KPER / KC)         // 64
#define ARGMAX_CHUNKS 8
#define ARGMAX_ROWS   1024      // 8 * 1024 = 8192 rows

typedef __attribute__((ext_vector_type(8)))  __bf16   bf16x8;
typedef __attribute__((ext_vector_type(16))) __bf16   v16bf;
typedef __attribute__((ext_vector_type(8)))  float    v8f;
typedef __attribute__((ext_vector_type(4)))  float    f32x4;
typedef __attribute__((ext_vector_type(2)))  unsigned u32x2;

// ---- WMMA fragment loaders (wave32, CDNA5 16-bit layouts) -------------------
// A 16x32 bf16: lanes 0-15 hold row M=lane, K = 0..7 (v0-3) and 16..23 (v4-7);
//               lanes 16-31 hold K = 8..15 and 24..31 of the same rows.
__device__ __forceinline__ v16bf ld_afrag(const __bf16* s, int rowBase, int kk, int lane) {
  int half = lane >> 4;
  int r    = rowBase + (lane & 15);
  const __bf16* base = s + r * LDST + kk + half * 8;
  bf16x8 a0 = *(const bf16x8*)(base);        // K = kk + half*8 .. +7
  bf16x8 a1 = *(const bf16x8*)(base + 16);   // K = kk + 16 + half*8 .. +7
  return __builtin_shufflevector(a0, a1, 0,1,2,3,4,5,6,7,8,9,10,11,12,13,14,15);
}

// B 32x16 bf16: lanes 0-15 hold column N=lane, K = 0..7 (v0-3) and 16..23 (v4-7);
//               lanes 16-31 hold K = 8..15 and 24..31. B[k][j] = logW[j][k] -> row j of LDS chunk.
__device__ __forceinline__ v16bf ld_bfrag(const __bf16* s, int colBase, int kk, int lane) {
  int half = lane >> 4;
  int j    = colBase + (lane & 15);
  const __bf16* base = s + j * LDST + kk + half * 8;
  bf16x8 b0 = *(const bf16x8*)(base);
  bf16x8 b1 = *(const bf16x8*)(base + 16);
  return __builtin_shufflevector(b0, b1, 0,1,2,3,4,5,6,7,8,9,10,11,12,13,14,15);
}

__device__ __forceinline__ v8f wmma_bf16(v16bf a, v16bf b, v8f c) {
  return __builtin_amdgcn_wmma_f32_16x16x32_bf16(
      /*neg_a=*/false, a, /*neg_b=*/false, b,
      /*c_mod=*/(short)0, c, /*reuse_a=*/false, /*reuse_b=*/false);
}

// ---- packed bf16 staging helpers -------------------------------------------
__device__ __forceinline__ unsigned pack2_bf16(float a, float b) {
  unsigned short ua = __builtin_bit_cast(unsigned short, (__bf16)a);
  unsigned short ub = __builtin_bit_cast(unsigned short, (__bf16)b);
  return ((unsigned)ub << 16) | (unsigned)ua;
}
__device__ __forceinline__ void store_pack4(__bf16* p, float a, float b, float c, float d) {
  u32x2 u;
  u.x = pack2_bf16(a, b);
  u.y = pack2_bf16(c, d);
  *(u32x2*)p = u;               // one ds_store_b64
}
__device__ __forceinline__ float bf16_rt(float x) {   // round-trip through bf16
  return (float)(__bf16)x;
}

// ---- Kernel 1: K-split, double-buffered, software-pipelined WMMA GEMM -------
// block (p, ks) accumulates partial (trace - sum) of M over K in [ks*KPER, (ks+1)*KPER)
// Body order per chunk: sync -> convert/store next chunk -> issue prefetch -> WMMA,
// so prefetched global loads get a full WMMA loop of latency before being drained.
__global__ void __launch_bounds__(256)
entropy_gemm_kernel(const float* __restrict__ W, float* __restrict__ per_proj_part) {
  __shared__ __bf16 sWhi[2][CHELEM];
  __shared__ __bf16 sWlo[2][CHELEM];
  __shared__ __bf16 sLhi[2][CHELEM];
  __shared__ __bf16 sLlo[2][CHELEM];
  __shared__ float  red[256];

  const int p    = blockIdx.x;
  const int ks   = blockIdx.y;
  const int t    = threadIdx.x;
  const int lane = t & 31;
  const int wave = t >> 5;
  const int mi     = wave & 3;            // C-tile row  (0..3)
  const int njBase = (wave >> 2) * 2;     // C-tile cols {njBase, njBase+1}

  v8f acc0 = {};
  v8f acc1 = {};

  const float* Wp = W + (size_t)p * NUM_GROUPS * IN_DIM + (size_t)ks * KPER;

  // Per-thread staging: 4 float4 per chunk; q = t + j*256 -> row q>>4, col (q&15)*4.
  // Consecutive t -> consecutive 16B segments => fully coalesced global_load_b128.
  f32x4 pre[4];

  #define LOAD_PRE(kn)                                                     \
    _Pragma("unroll")                                                      \
    for (int j = 0; j < 4; j++) {                                          \
      int q = t + j * 256;                                                 \
      int r = q >> 4, c = (q & 15) * 4;                                    \
      pre[j] = *(const f32x4*)(Wp + (size_t)r * IN_DIM + (kn) + c);        \
    }

  #define CONVERT_STORE(buf)                                               \
    _Pragma("unroll")                                                      \
    for (int j = 0; j < 4; j++) {                                          \
      int q = t + j * 256;                                                 \
      int r = q >> 4, c = (q & 15) * 4;                                    \
      f32x4 v = pre[j];                                                    \
      float l0 = __logf(fmaxf(v.x, EPSV));                                 \
      float l1 = __logf(fmaxf(v.y, EPSV));                                 \
      float l2 = __logf(fmaxf(v.z, EPSV));                                 \
      float l3 = __logf(fmaxf(v.w, EPSV));                                 \
      int o = r * LDST + c;                                                \
      store_pack4(&sWhi[buf][o], v.x, v.y, v.z, v.w);                      \
      store_pack4(&sWlo[buf][o], v.x - bf16_rt(v.x), v.y - bf16_rt(v.y),   \
                                 v.z - bf16_rt(v.z), v.w - bf16_rt(v.w));  \
      store_pack4(&sLhi[buf][o], l0, l1, l2, l3);                          \
      store_pack4(&sLlo[buf][o], l0 - bf16_rt(l0), l1 - bf16_rt(l1),       \
                                 l2 - bf16_rt(l2), l3 - bf16_rt(l3));      \
    }

  // Prologue: stage chunk 0 into buf 0; issue loads for chunk 1.
  LOAD_PRE(0)
  CONVERT_STORE(0)
  LOAD_PRE(KC)

  for (int chunk = 0; chunk < NCHUNK; chunk++) {
    const int cur = chunk & 1;
    __syncthreads();
    // Barrier guarantees: (a) buf[cur] stores (from prev iter / prologue) visible,
    // (b) all reads of buf[1-cur] from prev iter finished -> safe to overwrite now.

    if (chunk + 1 < NCHUNK) {
      // Drain loads for chunk+1 (issued last iteration, hidden behind its WMMAs),
      // convert into the idle buffer.
      CONVERT_STORE(1 - cur)
      if (chunk + 2 < NCHUNK) {
        LOAD_PRE((chunk + 2) * KC)   // in flight across the WMMAs below + barrier
      }
    }

    #pragma unroll
    for (int kk = 0; kk < KC; kk += 32) {
      v16bf ahi = ld_afrag(sWhi[cur], mi * 16, kk, lane);
      v16bf alo = ld_afrag(sWlo[cur], mi * 16, kk, lane);
      v16bf b0h = ld_bfrag(sLhi[cur], njBase * 16,       kk, lane);
      v16bf b0l = ld_bfrag(sLlo[cur], njBase * 16,       kk, lane);
      v16bf b1h = ld_bfrag(sLhi[cur], (njBase + 1) * 16, kk, lane);
      v16bf b1l = ld_bfrag(sLlo[cur], (njBase + 1) * 16, kk, lane);
      // M ~= Whi*Lhi + Whi*Llo + Wlo*Lhi  (lo*lo dropped: ~2^-16 relative)
      acc0 = wmma_bf16(ahi, b0h, acc0);
      acc0 = wmma_bf16(ahi, b0l, acc0);
      acc0 = wmma_bf16(alo, b0h, acc0);
      acc1 = wmma_bf16(ahi, b1h, acc1);
      acc1 = wmma_bf16(ahi, b1l, acc1);
      acc1 = wmma_bf16(alo, b1h, acc1);
    }
  }

  // Epilogue: per-lane (diag - all) using C layout:
  // lanes 0-15:  element (M=v,   N=lane)    ; lanes 16-31: (M=v+8, N=lane-16)
  float part = 0.f;
  const int n_in_tile = lane & 15;
  const int m_off     = (lane >= 16) ? 8 : 0;
  #pragma unroll
  for (int v = 0; v < 8; v++) {
    float c0 = acc0[v];
    float c1 = acc1[v];
    part -= c0 + c1;
    int m = v + m_off;
    if (m == n_in_tile) {
      if (mi == njBase)     part += c0;
      if (mi == njBase + 1) part += c1;
    }
  }

  red[t] = part;
  __syncthreads();
  #pragma unroll
  for (int s = 128; s > 0; s >>= 1) {
    if (t < s) red[t] += red[t + s];
    __syncthreads();
  }
  if (t == 0) per_proj_part[ks * NUM_PROJ + p] = red[0];

  #undef LOAD_PRE
  #undef CONVERT_STORE
}

// ---- Kernel 1b: deterministic combine of K-segment partials -----------------
__global__ void perproj_combine_kernel(const float* __restrict__ part,
                                       float* __restrict__ per_proj) {
  int p = blockIdx.x * 128 + threadIdx.x;
  if (p >= NUM_PROJ) return;
  float s = 0.f;
  #pragma unroll
  for (int k = 0; k < KSPLIT; k++) s += part[k * NUM_PROJ + p];
  per_proj[p] = s;
}

// ---- Kernel 2a: zero the column-sum accumulator -----------------------------
__global__ void zero_colsum_kernel(int* __restrict__ colsum) {
  int c = blockIdx.x * 256 + threadIdx.x;
  if (c < NUM_CLASSES) colsum[c] = 0;
}

// ---- Kernel 2b: valid[c] via column sums of prototype_class_identity --------
__global__ void colsum_kernel(const int* __restrict__ proto, int* __restrict__ colsum,
                              int rowsPerBlock) {
  int c = blockIdx.x * 256 + threadIdx.x;
  if (c >= NUM_CLASSES) return;
  int r0 = blockIdx.y * rowsPerBlock;
  int r1 = r0 + rowsPerBlock;
  if (r1 > NUM_PROTOS) r1 = NUM_PROTOS;
  int s = 0;
  for (int r = r0; r < r1; r++) s += proto[(size_t)r * NUM_CLASSES + c];
  atomicAdd(&colsum[c], s);
}

// ---- Kernel 3a: partial column argmax of group_class_identity ---------------
__global__ void argmax_part_kernel(const int* __restrict__ gci,
                                   int* __restrict__ pval, int* __restrict__ pidx) {
  int c = blockIdx.x * 256 + threadIdx.x;
  if (c >= NUM_CLASSES) return;
  int r0 = blockIdx.y * ARGMAX_ROWS;
  int r1 = r0 + ARGMAX_ROWS;
  if (r1 > NUM_PG) r1 = NUM_PG;
  int best = -2147483647 - 1;
  int bi   = r0;
  for (int r = r0; r < r1; r++) {
    int v = gci[(size_t)r * NUM_CLASSES + c];
    if (v > best) { best = v; bi = r; }   // strict '>' keeps first occurrence
  }
  pval[blockIdx.y * NUM_CLASSES + c] = best;
  pidx[blockIdx.y * NUM_CLASSES + c] = bi;
}

// ---- Kernel 3b: combine partials -> proj_ids[c] = argmax // NUM_GROUPS ------
__global__ void argmax_combine_kernel(const int* __restrict__ pval,
                                      const int* __restrict__ pidx,
                                      int* __restrict__ proj_ids) {
  int c = blockIdx.x * 256 + threadIdx.x;
  if (c >= NUM_CLASSES) return;
  int best = -2147483647 - 1;
  int bi   = 0;
  for (int ch = 0; ch < ARGMAX_CHUNKS; ch++) {
    int v = pval[ch * NUM_CLASSES + c];
    if (v > best) { best = v; bi = pidx[ch * NUM_CLASSES + c]; }  // chunks in row order
  }
  proj_ids[c] = bi / NUM_GROUPS;
}

// ---- Kernel 4: masked gather-mean -> scalar ---------------------------------
__global__ void finalize_kernel(const float* __restrict__ per_proj,
                                const int* __restrict__ colsum,
                                const int* __restrict__ proj_ids,
                                float* __restrict__ out) {
  __shared__ float sred[256];
  __shared__ int   scnt[256];
  int t = threadIdx.x;
  float tot = 0.f;
  int   cnt = 0;
  for (int c = t; c < NUM_CLASSES; c += 256) {
    if (colsum[c] != 0) { tot += per_proj[proj_ids[c]]; cnt++; }
  }
  sred[t] = tot; scnt[t] = cnt;
  __syncthreads();
  #pragma unroll
  for (int s = 128; s > 0; s >>= 1) {
    if (t < s) { sred[t] += sred[t + s]; scnt[t] += scnt[t + s]; }
    __syncthreads();
  }
  if (t == 0) {
    // count = nvalid * G*(G-1); result = -(total / count)
    float count = (float)scnt[0] * (float)(NUM_GROUPS * (NUM_GROUPS - 1));
    out[0] = -(sred[0] / count);
  }
}

extern "C" void kernel_launch(void* const* d_in, const int* in_sizes, int n_in,
                              void* d_out, int out_size, void* d_ws, size_t ws_size,
                              hipStream_t stream) {
  (void)in_sizes; (void)n_in; (void)out_size; (void)ws_size;
  const float* W     = (const float*)d_in[0];   // [128, 64, 16384] f32
  const int*   proto = (const int*)d_in[1];     // [10000, 1000]   i32
  const int*   gci   = (const int*)d_in[2];     // [8192, 1000]    i32
  float*       out   = (float*)d_out;           // scalar f32

  // workspace layout (4-byte units)
  float* per_proj = (float*)d_ws;                         // 128 f32
  float* pp_part  = per_proj + NUM_PROJ;                  // KSPLIT*128 f32
  int*   colsum   = (int*)(pp_part + KSPLIT * NUM_PROJ);  // 1000 i32
  int*   proj_ids = colsum + NUM_CLASSES;                 // 1000 i32
  int*   pval     = proj_ids + NUM_CLASSES;               // 8*1000 i32
  int*   pidx     = pval + ARGMAX_CHUNKS * NUM_CLASSES;   // 8*1000 i32

  entropy_gemm_kernel<<<dim3(NUM_PROJ, KSPLIT), 256, 0, stream>>>(W, pp_part);
  perproj_combine_kernel<<<1, 128, 0, stream>>>(pp_part, per_proj);
  zero_colsum_kernel<<<(NUM_CLASSES + 255) / 256, 256, 0, stream>>>(colsum);
  colsum_kernel<<<dim3((NUM_CLASSES + 255) / 256, 16), 256, 0, stream>>>(proto, colsum, 625);
  argmax_part_kernel<<<dim3((NUM_CLASSES + 255) / 256, ARGMAX_CHUNKS), 256, 0, stream>>>(gci, pval, pidx);
  argmax_combine_kernel<<<(NUM_CLASSES + 255) / 256, 256, 0, stream>>>(pval, pidx, proj_ids);
  finalize_kernel<<<1, 256, 0, stream>>>(per_proj, colsum, proj_ids, out);
}
